// EfficientDecoderBlock_83580063580455
// MI455X (gfx1250) — matrix operand order repaired and checked by reading, not compiled, not run on hardware
//
#include <hip/hip_runtime.h>
#include <hip/hip_bf16.h>
#include <math.h>

#define T_  4096
#define B_  8
#define D_  512
#define H_  8
#define BS_ 64
#define NB_ 64
#define DH_ 64
#define MTOT (T_*B_)

typedef _Float16 h16;
typedef _Float16 v8h  __attribute__((ext_vector_type(8)));
typedef _Float16 v16h __attribute__((ext_vector_type(16)));
typedef float    v8f  __attribute__((ext_vector_type(8)));
typedef int      v4i  __attribute__((ext_vector_type(4)));

static __device__ __forceinline__ v16h cat8(v8h lo, v8h hi) {
  return __builtin_shufflevector(lo, hi, 0,1,2,3,4,5,6,7,8,9,10,11,12,13,14,15);
}

// ---- CDNA5 async global->LDS copy (ASYNCcnt path), 16 bytes per lane ----
static __device__ __forceinline__ void async_cp16(const h16* g, h16* l) {
#if __has_builtin(__builtin_amdgcn_global_load_async_to_lds_b128)
  __builtin_amdgcn_global_load_async_to_lds_b128(
      (__attribute__((address_space(1))) v4i*)(void*)g,
      (__attribute__((address_space(3))) v4i*)(void*)l, 0, 0);
#else
  *(uint4*)l = *(const uint4*)g;
#endif
}
static __device__ __forceinline__ void wait_async() {
#if __has_builtin(__builtin_amdgcn_global_load_async_to_lds_b128)
#if __has_builtin(__builtin_amdgcn_s_wait_asynccnt)
  __builtin_amdgcn_s_wait_asynccnt(0);
#else
  asm volatile("s_wait_asynccnt 0x0" ::: "memory");
#endif
#endif
}

// ---- weight fp32 [K,N] -> f16 transposed [N,K] ----
__global__ void wt_f16_kernel(const float* __restrict__ W, h16* __restrict__ Wt,
                              int K, int N) {
  int idx = blockIdx.x * 256 + threadIdx.x;
  int k = idx / N, n = idx % N;
  Wt[(size_t)n * K + k] = (h16)W[idx];
}

// ---- depthwise conv3 + residual + BN (inference) ----
__global__ void conv_bn_kernel(const float* __restrict__ x, const float* __restrict__ cw,
                               const float* __restrict__ bg, const float* __restrict__ bb,
                               const float* __restrict__ bm, const float* __restrict__ bv,
                               float* __restrict__ x1, h16* __restrict__ x1h) {
  size_t idx = (size_t)blockIdx.x * 256 + threadIdx.x;
  int d = (int)(idx % D_);
  size_t t = idx / ((size_t)B_ * D_);
  float xc = x[idx];
  float xm = (t > 0)      ? x[idx - (size_t)B_*D_] : 0.f;
  float xp = (t < T_ - 1) ? x[idx + (size_t)B_*D_] : 0.f;
  float y = cw[d*3+0]*xm + cw[d*3+1]*xc + cw[d*3+2]*xp + xc;
  y = (y - bm[d]) * rsqrtf(bv[d] + 1e-5f) * bg[d] + bb[d];
  x1[idx] = y;
  x1h[idx] = (h16)y;
}

// ---- WMMA GEMM, 128x128 block tile, double-buffered async LDS ----
// C = act(A[M,K] @ Bt[N,K]^T + bias); f32 and/or f16 outputs.
__global__ __launch_bounds__(256)
void wmma_gemm_kernel(const h16* __restrict__ A, const h16* __restrict__ Bt,
                      const float* __restrict__ bias,
                      float* __restrict__ C, h16* __restrict__ Ch,
                      int N, int K, int act) {
  __shared__ __attribute__((aligned(32))) h16 As[2][128 * 32];
  __shared__ __attribute__((aligned(32))) h16 Bs[2][128 * 32];
  int tid = threadIdx.x;
  int w = tid >> 5, lane = tid & 31;
  int wm = (w & 3) * 32, wn = (w >> 2) * 64;
  int m0 = blockIdx.y * 128, n0 = blockIdx.x * 128;
  v8f acc[2][4] = {};
  int row = tid >> 1, koff = (tid & 1) * 16;
  int mlane = lane & 15;
  int kbA = (lane >> 4) * 16;
  int kbB = (lane >> 4) * 16;
  const h16* Ag = A  + (size_t)(m0 + row) * K + koff;
  const h16* Bg = Bt + (size_t)(n0 + row) * K + koff;
  h16* asd = &As[0][row * 32 + koff];
  h16* bsd = &Bs[0][row * 32 + koff];
  const int lbuf = 128 * 32;  // halfs per buffer

  // prologue: tile 0 into buffer 0
  async_cp16(Ag,     asd);
  async_cp16(Ag + 8, asd + 8);
  async_cp16(Bg,     bsd);
  async_cp16(Bg + 8, bsd + 8);

  int nk = K >> 5;
  for (int ki = 0; ki < nk; ++ki) {
    int cur = ki & 1;
    wait_async();
    __syncthreads();
    if (ki + 1 < nk) {  // stream next tile into the other buffer
      int nxt = cur ^ 1;
      const h16* an = Ag + (size_t)(ki + 1) * 32;
      const h16* bn = Bg + (size_t)(ki + 1) * 32;
      async_cp16(an,     asd + nxt * lbuf);
      async_cp16(an + 8, asd + nxt * lbuf + 8);
      async_cp16(bn,     bsd + nxt * lbuf);
      async_cp16(bn + 8, bsd + nxt * lbuf + 8);
    }
    v16h af[2], bf[4];
#pragma unroll
    for (int tm = 0; tm < 2; ++tm) {
      const h16* p = &As[cur][(wm + tm * 16 + mlane) * 32 + kbA];
      af[tm] = cat8(*(const v8h*)p, *(const v8h*)(p + 8));
    }
#pragma unroll
    for (int tn = 0; tn < 4; ++tn)
      bf[tn] = *(const v16h*)&Bs[cur][(wn + tn * 16 + mlane) * 32 + kbB];
#pragma unroll
    for (int tm = 0; tm < 2; ++tm)
#pragma unroll
      for (int tn = 0; tn < 4; ++tn)
        acc[tm][tn] = __builtin_amdgcn_wmma_f32_16x16x32_f16(
            false, af[tm], false, bf[tn], (short)0, acc[tm][tn], false, false);
    __syncthreads();
  }

  int mseg = (lane >> 4) * 8;
#pragma unroll
  for (int tm = 0; tm < 2; ++tm)
#pragma unroll
    for (int tn = 0; tn < 4; ++tn)
#pragma unroll
      for (int r = 0; r < 8; ++r) {
        int gm = m0 + wm + tm * 16 + mseg + r;
        int gn = n0 + wn + tn * 16 + mlane;
        float v = acc[tm][tn][r] + bias[gn];
        if (act) v = 0.5f * v * (1.0f + erff(v * 0.70710678118f));
        size_t o = (size_t)gm * N + gn;
        if (C)  C[o]  = v;
        if (Ch) Ch[o] = (h16)v;
      }
}

// ---- bucketed sparse attention: one block per (b, head, bucket) ----
__global__ __launch_bounds__(256)
void attn_kernel(const h16* __restrict__ qh, const h16* __restrict__ kh,
                 const h16* __restrict__ vh,
                 const int* __restrict__ t_length, const int* __restrict__ rand_idx,
                 h16* __restrict__ attnh) {
  __shared__ __attribute__((aligned(32))) h16   Qs[64 * 64];
  __shared__ __attribute__((aligned(32))) h16   Ks[128 * 64];
  __shared__ __attribute__((aligned(32))) h16   Vt[64 * 128];   // [dh][key]
  __shared__ __attribute__((aligned(32))) float Sc[64 * 128];
  __shared__ __attribute__((aligned(32))) h16   Ps[64 * 128];
  int bx = blockIdx.x;
  int nb = bx % NB_;
  int hd = (bx / NB_) % H_;
  int b  = bx / (NB_ * H_);
  int tid = threadIdx.x;
  int rnb  = rand_idx[nb];
  int tlen = t_length[b];

  { // Q tile [64][64] via async-to-LDS
    int q = tid >> 2, doff = (tid & 3) * 16;
    int t = nb * BS_ + q;
    const h16* g = qh + ((size_t)t * B_ + b) * D_ + hd * DH_ + doff;
    h16* s = Qs + q * 64 + doff;
    async_cp16(g, s);
    async_cp16(g + 8, s + 8);
  }
  { // K tile [128][64] (own + random bucket) via async-to-LDS
    int key = tid >> 1, doff = (tid & 1) * 32;
    int t = (key < BS_) ? nb * BS_ + key : rnb * BS_ + (key - BS_);
    const h16* g = kh + ((size_t)t * B_ + b) * D_ + hd * DH_ + doff;
    h16* s = Ks + key * 64 + doff;
    async_cp16(g,      s);
    async_cp16(g + 8,  s + 8);
    async_cp16(g + 16, s + 16);
    async_cp16(g + 24, s + 24);
  }
  { // V transposed into Vt[dh][key] (register path: needs transpose)
    int key = tid >> 1, doff = (tid & 1) * 32;
    int t = (key < BS_) ? nb * BS_ + key : rnb * BS_ + (key - BS_);
    const uint4* g = (const uint4*)(vh + ((size_t)t * B_ + b) * D_ + hd * DH_ + doff);
    union { uint4 u[4]; h16 f[32]; } tmp;
    tmp.u[0] = g[0]; tmp.u[1] = g[1]; tmp.u[2] = g[2]; tmp.u[3] = g[3];
#pragma unroll
    for (int i = 0; i < 32; ++i) Vt[(doff + i) * 128 + key] = tmp.f[i];
  }
  wait_async();
  __syncthreads();

  int w = tid >> 5, lane = tid & 31;
  int mlane = lane & 15, kbA = (lane >> 4) * 16, kbB = (lane >> 4) * 16;
  int mseg = (lane >> 4) * 8;

  { // scores = Q @ K^T, scale + mask, to LDS f32
    int qr0 = (w & 3) * 16, kc0 = (w >> 2) * 64;
    v8f s[4] = {};
#pragma unroll
    for (int d0 = 0; d0 < 64; d0 += 32) {
      const h16* ap = Qs + (qr0 + mlane) * 64 + d0 + kbA;
      v16h af = cat8(*(const v8h*)ap, *(const v8h*)(ap + 8));
#pragma unroll
      for (int tn = 0; tn < 4; ++tn) {
        v16h bf = *(const v16h*)(Ks + (kc0 + tn * 16 + mlane) * 64 + d0 + kbB);
        s[tn] = __builtin_amdgcn_wmma_f32_16x16x32_f16(
            false, af, false, bf, (short)0, s[tn], false, false);
      }
    }
#pragma unroll
    for (int tn = 0; tn < 4; ++tn)
#pragma unroll
      for (int r = 0; r < 8; ++r) {
        int q  = qr0 + mseg + r;
        int kc = kc0 + tn * 16 + mlane;
        int kp = (kc < BS_) ? nb * BS_ + kc : rnb * BS_ + (kc - BS_);
        float v = s[tn][r] * 0.125f;
        if (kp >= tlen) v = -1e9f;
        Sc[q * 128 + kc] = v;
      }
  }
  __syncthreads();

  if (tid < 64) { // row softmax -> f16 probabilities
    float mx = -3.4e38f;
    for (int c = 0; c < 128; ++c) mx = fmaxf(mx, Sc[tid * 128 + c]);
    float ssum = 0.f;
    for (int c = 0; c < 128; ++c) {
      float e = expf(Sc[tid * 128 + c] - mx);
      Sc[tid * 128 + c] = e;
      ssum += e;
    }
    float inv = 1.0f / ssum;
    for (int c = 0; c < 128; ++c) Ps[tid * 128 + c] = (h16)(Sc[tid * 128 + c] * inv);
  }
  __syncthreads();

  { // O = P @ V
    int qr0 = (w & 3) * 16, dc0 = (w >> 2) * 32;
    v8f o[2] = {};
#pragma unroll
    for (int k0 = 0; k0 < 128; k0 += 32) {
      const h16* ap = Ps + (qr0 + mlane) * 128 + k0 + kbA;
      v16h af = cat8(*(const v8h*)ap, *(const v8h*)(ap + 8));
#pragma unroll
      for (int tn = 0; tn < 2; ++tn) {
        v16h bf = *(const v16h*)(Vt + (dc0 + tn * 16 + mlane) * 128 + k0 + kbB);
        o[tn] = __builtin_amdgcn_wmma_f32_16x16x32_f16(
            false, af, false, bf, (short)0, o[tn], false, false);
      }
    }
#pragma unroll
    for (int tn = 0; tn < 2; ++tn)
#pragma unroll
      for (int r = 0; r < 8; ++r) {
        int t  = nb * BS_ + qr0 + mseg + r;
        int dd = hd * DH_ + dc0 + tn * 16 + mlane;
        attnh[((size_t)t * B_ + b) * D_ + dd] = (h16)o[tn][r];
      }
  }
}

// ---- residual + LayerNorm (D=512, 1 block per row) ----
__global__ __launch_bounds__(256)
void ln_kernel(const float* __restrict__ a, const float* __restrict__ resid,
               const float* __restrict__ g, const float* __restrict__ bb,
               float* __restrict__ outp, h16* __restrict__ outh) {
  __shared__ float red[256];
  int row = blockIdx.x, tid = threadIdx.x;
  size_t base = (size_t)row * D_;
  float v0 = a[base + tid]       + resid[base + tid];
  float v1 = a[base + tid + 256] + resid[base + tid + 256];
  red[tid] = v0 + v1; __syncthreads();
  for (int o = 128; o > 0; o >>= 1) { if (tid < o) red[tid] += red[tid + o]; __syncthreads(); }
  float mean = red[0] * (1.0f / 512.0f);
  __syncthreads();
  float d0 = v0 - mean, d1 = v1 - mean;
  red[tid] = d0 * d0 + d1 * d1; __syncthreads();
  for (int o = 128; o > 0; o >>= 1) { if (tid < o) red[tid] += red[tid + o]; __syncthreads(); }
  float inv = rsqrtf(red[0] * (1.0f / 512.0f) + 1e-5f);
  float o0 = d0 * inv * g[tid]       + bb[tid];
  float o1 = d1 * inv * g[tid + 256] + bb[tid + 256];
  outp[base + tid]       = o0;
  outp[base + tid + 256] = o1;
  if (outh) { outh[base + tid] = (h16)o0; outh[base + tid + 256] = (h16)o1; }
}

extern "C" void kernel_launch(void* const* d_in, const int* in_sizes, int n_in,
                              void* d_out, int out_size, void* d_ws, size_t ws_size,
                              hipStream_t stream) {
  (void)in_sizes; (void)n_in; (void)out_size; (void)ws_size;
  const float* x        = (const float*)d_in[0];
  const int*   t_length = (const int*)d_in[1];
  const int*   rand_idx = (const int*)d_in[2];
  const float* conv_w   = (const float*)d_in[3];
  const float* bn_g = (const float*)d_in[4];
  const float* bn_b = (const float*)d_in[5];
  const float* bn_m = (const float*)d_in[6];
  const float* bn_v = (const float*)d_in[7];
  const float* wq = (const float*)d_in[8];  const float* bq = (const float*)d_in[9];
  const float* wk = (const float*)d_in[10]; const float* bk = (const float*)d_in[11];
  const float* wv = (const float*)d_in[12]; const float* bv = (const float*)d_in[13];
  const float* wo = (const float*)d_in[14]; const float* bo = (const float*)d_in[15];
  const float* ln1_g = (const float*)d_in[16]; const float* ln1_b = (const float*)d_in[17];
  const float* w1 = (const float*)d_in[18]; const float* b1 = (const float*)d_in[19];
  const float* w2 = (const float*)d_in[20]; const float* b2 = (const float*)d_in[21];
  const float* ln2_g = (const float*)d_in[22]; const float* ln2_b = (const float*)d_in[23];

  char* ws = (char*)d_ws;
  const size_t MB = 1ull << 20;
  h16* wqt = (h16*)(ws + 0 * 512 * 1024);
  h16* wkt = (h16*)(ws + 1 * 512 * 1024);
  h16* wvt = (h16*)(ws + 2 * 512 * 1024);
  h16* wot = (h16*)(ws + 3 * 512 * 1024);
  h16* w1t = (h16*)(ws + 2 * MB);
  h16* w2t = (h16*)(ws + 3 * MB);
  float* x1    = (float*)(ws + 4 * MB);    // 64MB, later reused in-place as x2
  h16*   x1h   = (h16*)(ws + 68 * MB);     // 32MB
  h16*   qh    = (h16*)(ws + 100 * MB);    // 32MB
  h16*   khb   = (h16*)(ws + 132 * MB);    // 32MB
  h16*   vhb   = (h16*)(ws + 164 * MB);    // 32MB
  h16*   attnh = (h16*)(ws + 196 * MB);    // 32MB
  float* oproj = (float*)(ws + 100 * MB);  // reuse qh/kh after attention
  float* x2    = x1;                       // in-place residual+LN
  h16*   x2h   = (h16*)(ws + 164 * MB);    // reuse vh
  h16*   hid   = (h16*)(ws + 100 * MB);    // 64MB, reuse oproj after ln1
  float* mlp   = (float*)(ws + 164 * MB);  // 64MB, reuse x2h+attnh

  wt_f16_kernel<<<(512 * 512) / 256, 256, 0, stream>>>(wq, wqt, 512, 512);
  wt_f16_kernel<<<(512 * 512) / 256, 256, 0, stream>>>(wk, wkt, 512, 512);
  wt_f16_kernel<<<(512 * 512) / 256, 256, 0, stream>>>(wv, wvt, 512, 512);
  wt_f16_kernel<<<(512 * 512) / 256, 256, 0, stream>>>(wo, wot, 512, 512);
  wt_f16_kernel<<<(512 * 1024) / 256, 256, 0, stream>>>(w1, w1t, 512, 1024);
  wt_f16_kernel<<<(1024 * 512) / 256, 256, 0, stream>>>(w2, w2t, 1024, 512);

  conv_bn_kernel<<<(T_ * B_ * D_) / 256, 256, 0, stream>>>(
      x, conv_w, bn_g, bn_b, bn_m, bn_v, x1, x1h);

  dim3 g512(512 / 128, MTOT / 128);
  dim3 g1024(1024 / 128, MTOT / 128);
  wmma_gemm_kernel<<<g512, 256, 0, stream>>>(x1h, wqt, bq, nullptr, qh,  512, 512, 0);
  wmma_gemm_kernel<<<g512, 256, 0, stream>>>(x1h, wkt, bk, nullptr, khb, 512, 512, 0);
  wmma_gemm_kernel<<<g512, 256, 0, stream>>>(x1h, wvt, bv, nullptr, vhb, 512, 512, 0);

  attn_kernel<<<B_ * H_ * NB_, 256, 0, stream>>>(qh, khb, vhb, t_length, rand_idx, attnh);

  wmma_gemm_kernel<<<g512, 256, 0, stream>>>(attnh, wot, bo, oproj, nullptr, 512, 512, 0);
  ln_kernel<<<MTOT, 256, 0, stream>>>(oproj, x1, ln1_g, ln1_b, x2, x2h);

  wmma_gemm_kernel<<<g1024, 256, 0, stream>>>(x2h, w1t, b1, nullptr, hid, 1024, 512, 1);
  wmma_gemm_kernel<<<g512, 256, 0, stream>>>(hid, w2t, b2, mlp, nullptr, 512, 1024, 0);
  ln_kernel<<<MTOT, 256, 0, stream>>>(mlp, x2, ln2_g, ln2_b, (float*)d_out, nullptr);
}
